// NeRFAttention_54992761258555
// MI455X (gfx1250) — matrix operand order repaired and checked by reading, not compiled, not run on hardware
//
#include <hip/hip_runtime.h>
#include <hip/hip_bf16.h>

typedef __attribute__((ext_vector_type(16))) _Float16 v16h;
typedef __attribute__((ext_vector_type(16))) __bf16   v16bf;
typedef __attribute__((ext_vector_type(8)))  float    v8f;
typedef __attribute__((ext_vector_type(4)))  float    v4f;

#define N_TOK  8192
#define N_EMB  64
#define N_POS  16

// ---- WMMA fragment loaders (wave32; contiguous k order shared by A and B) ----
// A (16x32, 16-bit): lane m=L&15; elem j -> k=(L>>4)*16+j (16 contiguous halfs)
__device__ inline v16h frag_a_f16(const _Float16* M, int row, int kb, int lane) {
  const _Float16* p = M + (size_t)row * N_EMB + kb + ((lane >> 4) << 4);
  v16h f;
#pragma unroll
  for (int j = 0; j < 16; ++j) f[j] = p[j];
  return f;
}
// B (32x16, 16-bit): lane n=L&15; elem j -> k=(L>>4)*16+j (16 contiguous halfs)
__device__ inline v16h frag_b_f16(const _Float16* M, int col, int kb, int lane) {
  const _Float16* p = M + (size_t)col * N_EMB + kb + ((lane >> 4) << 4);
  v16h f;
#pragma unroll
  for (int j = 0; j < 16; ++j) f[j] = p[j];
  return f;
}
// A fragment for P@f: read exp-scores (f32, row width 8192) and convert to bf16
__device__ inline v16bf frag_a_bf16_e(const float* e, int row, int nb, int lane) {
  const float* p = e + (size_t)row * N_TOK + nb + ((lane >> 4) << 4);
  v16bf f;
#pragma unroll
  for (int j = 0; j < 16; ++j) f[j] = (__bf16)p[j];
  return f;
}
// B fragment from transposed features ft[64][8192] (bf16)
__device__ inline v16bf frag_b_bf16(const __bf16* ftp, int col, int nb, int lane) {
  const __bf16* p = ftp + (size_t)col * N_TOK + nb + ((lane >> 4) << 4);
  v16bf f;
#pragma unroll
  for (int j = 0; j < 16; ++j) f[j] = p[j];
  return f;
}

// ---- kernel 1: norm_scale = scale @ Ws + bs ----
__global__ void nerf_scale_kernel(const float* __restrict__ scale,
                                  const float* __restrict__ Ws,
                                  const float* __restrict__ bs,
                                  float* __restrict__ ns) {
  if (threadIdx.x == 0)
    ns[0] = scale[0] * Ws[0] + scale[1] * Ws[1] + scale[2] * Ws[2] + bs[0];
}

// ---- tiny MLP (16 -> relu 32 -> 64 -> LayerNorm) per token, result scaled -> f16 ----
__device__ inline void mlp_row(const float* pos,
                               const float* __restrict__ W1, const float* __restrict__ b1,
                               const float* __restrict__ W2, const float* __restrict__ b2,
                               const float* __restrict__ g,  const float* __restrict__ b,
                               float sc, _Float16* __restrict__ dst) {
  float h[32];
#pragma unroll
  for (int j = 0; j < 32; ++j) {
    float s = b1[j];
#pragma unroll
    for (int i = 0; i < N_POS; ++i) s = fmaf(pos[i], W1[i * 32 + j], s);
    h[j] = s > 0.f ? s : 0.f;
  }
  float e[64], mean = 0.f;
#pragma unroll
  for (int c = 0; c < 64; ++c) {
    float s = b2[c];
#pragma unroll
    for (int j = 0; j < 32; ++j) s = fmaf(h[j], W2[j * 64 + c], s);
    e[c] = s; mean += s;
  }
  mean *= (1.f / 64.f);
  float var = 0.f;
#pragma unroll
  for (int c = 0; c < 64; ++c) { float d = e[c] - mean; var = fmaf(d, d, var); }
  float inv = rsqrtf(var * (1.f / 64.f) + 1e-5f);
#pragma unroll
  for (int c = 0; c < 64; ++c)
    dst[c] = (_Float16)((fmaf((e[c] - mean) * inv, g[c], b[c])) * sc);
}

// ---- kernel 2: Q,K (f16, scale folded into Q) + transpose features to bf16 ----
__global__ void nerf_qk_kernel(const float* __restrict__ rays, const float* __restrict__ feat,
                               const float* __restrict__ Wq1, const float* __restrict__ bq1,
                               const float* __restrict__ Wq2, const float* __restrict__ bq2,
                               const float* __restrict__ gq,  const float* __restrict__ bqln,
                               const float* __restrict__ Wk1, const float* __restrict__ bk1,
                               const float* __restrict__ Wk2, const float* __restrict__ bk2,
                               const float* __restrict__ gk,  const float* __restrict__ bkln,
                               const float* __restrict__ ns,
                               _Float16* __restrict__ Qh, _Float16* __restrict__ Kh,
                               __bf16* __restrict__ ft) {
  int r = blockIdx.x * blockDim.x + threadIdx.x;
  if (r >= N_TOK) return;
  float pos[N_POS];
#pragma unroll
  for (int i = 0; i < N_POS; ++i) pos[i] = rays[(size_t)r * N_POS + i];
  // features transpose: ft[c][r] (bf16) for contiguous B-fragments in P@f
#pragma unroll 4
  for (int c = 0; c < 64; ++c)
    ft[(size_t)c * N_TOK + r] = (__bf16)feat[(size_t)r * 64 + c];
  float qs = ns[0] * 0.125f;  // norm_scale / sqrt(64)
  mlp_row(pos, Wq1, bq1, Wq2, bq2, gq, bqln, qs,  Qh + (size_t)r * N_EMB);
  mlp_row(pos, Wk1, bk1, Wk2, bk2, gk, bkln, 1.f, Kh + (size_t)r * N_EMB);
}

// ---- kernel 3: e = exp(Q K^T) via f16 WMMA; write e, compute deterministic row sums ----
// grid = 512 blocks (one per 16-row tile), 128 threads = 4 waves; wave handles 128 col-tiles.
__global__ void nerf_score_kernel(const _Float16* __restrict__ Qh,
                                  const _Float16* __restrict__ Kh,
                                  float* __restrict__ attn,
                                  float* __restrict__ row_sum) {
  const int lane = threadIdx.x & 31;
  const int wave = threadIdx.x >> 5;
  const int rbase = blockIdx.x * 16;
  const v16h a0 = frag_a_f16(Qh, rbase + (lane & 15), 0, lane);
  const v16h a1 = frag_a_f16(Qh, rbase + (lane & 15), 32, lane);
  float psum[8] = {0.f, 0.f, 0.f, 0.f, 0.f, 0.f, 0.f, 0.f};
  for (int it = 0; it < 128; ++it) {
    const int cbase = (it * 4 + wave) * 16;
    v16h b0 = frag_b_f16(Kh, cbase + (lane & 15), 0, lane);
    v16h b1 = frag_b_f16(Kh, cbase + (lane & 15), 32, lane);
    v8f c = {};
    c = __builtin_amdgcn_wmma_f32_16x16x32_f16(false, a0, false, b0, (short)0, c, false, false);
    c = __builtin_amdgcn_wmma_f32_16x16x32_f16(false, a1, false, b1, (short)0, c, false, false);
#pragma unroll
    for (int j = 0; j < 8; ++j) {
      float ev = __expf(c[j]);
      int row = rbase + j + ((lane >> 4) << 3);
      attn[(size_t)row * N_TOK + cbase + (lane & 15)] = ev;
      psum[j] += ev;
    }
  }
  // reduce over the 16 lanes sharing a row (xor masks < 16 stay within each half-wave)
#pragma unroll
  for (int j = 0; j < 8; ++j) {
    float v = psum[j];
    v += __shfl_xor(v, 8, 32);
    v += __shfl_xor(v, 4, 32);
    v += __shfl_xor(v, 2, 32);
    v += __shfl_xor(v, 1, 32);
    psum[j] = v;
  }
  __shared__ float lsum[4][16];
  if ((lane & 15) == 0) {
#pragma unroll
    for (int j = 0; j < 8; ++j) lsum[wave][j + ((lane >> 4) << 3)] = psum[j];
  }
  __syncthreads();
  if (threadIdx.x < 16) {
    // fixed summation order -> deterministic
    float s = lsum[0][threadIdx.x] + lsum[1][threadIdx.x] +
              lsum[2][threadIdx.x] + lsum[3][threadIdx.x];
    row_sum[rbase + threadIdx.x] = s;
  }
}

// ---- kernel 4: fused attn_out = (e @ f) * inv_sum (bf16 WMMA) + in-place normalize of e ----
// grid = 512 blocks (one per 16-row tile), 128 threads = 4 waves; wave w owns out cols [16w,16w+16)
__global__ void nerf_out_kernel(float* __restrict__ attn,
                                const __bf16* __restrict__ ft,
                                const float* __restrict__ row_sum,
                                float* __restrict__ out) {
  const int lane = threadIdx.x & 31;
  const int wave = threadIdx.x >> 5;
  const int rbase = blockIdx.x * 16;
  const int cbase = wave * 16;
  __shared__ float sinv[16];
  if (threadIdx.x < 16) sinv[threadIdx.x] = 1.f / row_sum[rbase + threadIdx.x];
  __syncthreads();
  v8f acc = {};
  for (int nb = 0; nb < N_TOK; nb += 32) {
    v16bf a = frag_a_bf16_e(attn, rbase + (lane & 15), nb, lane);
    v16bf b = frag_b_bf16(ft, cbase + (lane & 15), nb, lane);
    acc = __builtin_amdgcn_wmma_f32_16x16x32_bf16(false, a, false, b, (short)0, acc, false, false);
  }
#pragma unroll
  for (int j = 0; j < 8; ++j) {
    int m = j + ((lane >> 4) << 3);
    out[(size_t)(rbase + m) * 64 + cbase + (lane & 15)] = acc[j] * sinv[m];
  }
  // all reads of raw e for this (block-exclusive) row tile are done -> normalize in place
  __syncthreads();
  v4f* a4 = (v4f*)(attn + (size_t)rbase * N_TOK);
  for (int t = threadIdx.x; t < 16 * (N_TOK / 4); t += 128) {
    float iv = sinv[t >> 11];           // 2048 v4f per row
    v4f v = a4[t];
    v *= iv;
    __builtin_nontemporal_store(v, &a4[t]);  // final output, never re-read
  }
}

extern "C" void kernel_launch(void* const* d_in, const int* in_sizes, int n_in,
                              void* d_out, int out_size, void* d_ws, size_t ws_size,
                              hipStream_t stream) {
  const float* feat = (const float*)d_in[0];
  const float* rays = (const float*)d_in[1];
  const float* scl  = (const float*)d_in[2];
  const float* Wq1  = (const float*)d_in[3];
  const float* bq1  = (const float*)d_in[4];
  const float* Wq2  = (const float*)d_in[5];
  const float* bq2  = (const float*)d_in[6];
  const float* gq   = (const float*)d_in[7];
  const float* bqln = (const float*)d_in[8];
  const float* Wk1  = (const float*)d_in[9];
  const float* bk1  = (const float*)d_in[10];
  const float* Wk2  = (const float*)d_in[11];
  const float* bk2  = (const float*)d_in[12];
  const float* gk   = (const float*)d_in[13];
  const float* bkln = (const float*)d_in[14];
  const float* Ws   = (const float*)d_in[15];
  const float* bs   = (const float*)d_in[16];

  // workspace layout (~3.2 MB)
  char* ws = (char*)d_ws;
  float*    ns      = (float*)ws;                               // 16 B slot
  float*    row_sum = (float*)(ws + 16);                        // 32 KB
  _Float16* Qh      = (_Float16*)(ws + 16 + 32768);             // 1 MB
  _Float16* Kh      = Qh + (size_t)N_TOK * N_EMB;               // 1 MB
  __bf16*   ft      = (__bf16*)(Kh + (size_t)N_TOK * N_EMB);    // 1 MB

  float* out  = (float*)d_out;                 // attn_out: 8192*64 floats
  float* attn = out + (size_t)N_TOK * 64;      // attn_weights: 8192*8192 floats

  nerf_scale_kernel<<<1, 32, 0, stream>>>(scl, Ws, bs, ns);
  nerf_qk_kernel<<<N_TOK / 256, 256, 0, stream>>>(rays, feat,
      Wq1, bq1, Wq2, bq2, gq, bqln, Wk1, bk1, Wk2, bk2, gk, bkln, ns, Qh, Kh, ft);
  nerf_score_kernel<<<N_TOK / 16, 128, 0, stream>>>(Qh, Kh, attn, row_sum);
  nerf_out_kernel<<<N_TOK / 16, 128, 0, stream>>>(attn, ft, row_sum, out);
}
